// VanillaWAP_17660905521273
// MI455X (gfx1250) — compile-verified, matrix operand order checked by reading, not run on hardware
//
#include <hip/hip_runtime.h>
#include <hip/hip_bf16.h>
#include <math.h>

// ---------------------------------------------------------------------------
// Types for WMMA fragments (gfx1250, wave32)
// ---------------------------------------------------------------------------
typedef _Float16 v16h __attribute__((ext_vector_type(16)));
typedef _Float16 h8v  __attribute__((ext_vector_type(8)));
typedef float    v8f  __attribute__((ext_vector_type(8)));

#define CDIV(a, b) (((a) + (b) - 1) / (b))

// ---------------------------------------------------------------------------
// Model constants
// ---------------------------------------------------------------------------
#define BSZ   32
#define DIM   128
#define HF    8
#define WF    32
#define LLEN  256     // HF*WF
#define VOC   512
#define EMB   256
#define HID   256
#define MAXLEN 128

// Input index map (setup_inputs order: x, target, then params leaves in
// sorted-key order; each {'b','w'} dict sorted -> b before w; conv dicts
// {'b','beta','gamma','w'})
enum {
  IN_X = 0, IN_TARGET = 1,
  IN_Ccr_b = 2, IN_Ccr_w = 3, IN_Ccz_b = 4, IN_Ccz_w = 5,
  IN_Ua_b = 6, IN_Ua_w = 7,
  IN_Uhr_b = 8, IN_Uhr_w = 9, IN_Uhz_b = 10, IN_Uhz_w = 11,
  IN_Urh_b = 12, IN_Urh_w = 13,
  IN_Wa_b = 14, IN_Wa_w = 15,
  IN_Wc_b = 16, IN_Wc_w = 17, IN_Wh_b = 18, IN_Wh_w = 19,
  IN_Wo_b = 20, IN_Wo_w = 21,
  IN_Wyh_b = 22, IN_Wyh_w = 23, IN_Wyr_b = 24, IN_Wyr_w = 25,
  IN_Wyz_b = 26, IN_Wyz_w = 27,
  IN_c0_b = 28, IN_c0_beta = 29, IN_c0_gamma = 30, IN_c0_w = 31,
  IN_c1_b = 32, IN_c1_beta = 33, IN_c1_gamma = 34, IN_c1_w = 35,
  IN_c2_b = 36, IN_c2_beta = 37, IN_c2_gamma = 38, IN_c2_w = 39,
  IN_c3_b = 40, IN_c3_beta = 41, IN_c3_gamma = 42, IN_c3_w = 43,
  IN_emb = 44, IN_nu_w = 45
};

// ---------------------------------------------------------------------------
// WMMA GEMM:  C[M,N] (f32, row stride ldc) = A[M,K](f16) * W[N,K](f16)^T
//             + bias[N] (+ C if accumulate). One 16x16 tile per wave.
// Fragment loads follow the CDNA5 ISA 16-bit A 16x32 layout:
//   lane<16 : row = lane,     K in {0..7, 16..23}
//   lane>=16: row = lane-16,  K in {8..15, 24..31}
// B fragment mirrors A over the weight matrix stored (N,K) row-major.
// ---------------------------------------------------------------------------
__global__ void gemm_wmma_f16(const _Float16* __restrict__ A,
                              const _Float16* __restrict__ Wt,
                              const float* __restrict__ bias,
                              float* __restrict__ C,
                              int M, int N, int K, int ldc,
                              int accumulate, int ntiles) {
  int wave = blockIdx.x * (blockDim.x >> 5) + (threadIdx.x >> 5);
  if (wave >= ntiles) return;          // uniform per wave: EXEC stays all-ones
  int lane = threadIdx.x & 31;
  int ntN = N >> 4;
  int m0 = (wave / ntN) << 4;
  int n0 = (wave % ntN) << 4;
  int r  = lane & 15;
  int hi = lane >> 4;

  const _Float16* ap = A  + (size_t)(m0 + r) * K + hi * 8;
  const _Float16* bp = Wt + (size_t)(n0 + r) * K + hi * 8;

  v8f acc = {};
  for (int k = 0; k < K; k += 32) {
    union Frag { v16h v; h8v h[2]; };
    Frag fa, fb;
    fa.h[0] = *(const h8v*)(ap + k);
    fa.h[1] = *(const h8v*)(ap + k + 16);
    fb.h[0] = *(const h8v*)(bp + k);
    fb.h[1] = *(const h8v*)(bp + k + 16);
    acc = __builtin_amdgcn_wmma_f32_16x16x32_f16(
        /*neg_a=*/false, fa.v, /*neg_b=*/false, fb.v,
        /*c_mod=*/(short)0, acc, /*reuse_a=*/false, /*reuse_b=*/false);
  }

  int col = n0 + r;
  float bv = bias ? bias[col] : 0.0f;
#pragma unroll
  for (int j = 0; j < 8; ++j) {
    int row = m0 + j + (hi << 3);
    size_t idx = (size_t)row * (size_t)ldc + (size_t)col;
    float v = acc[j] + bv;
    if (accumulate) v += C[idx];
    C[idx] = v;
  }
}

// ---------------------------------------------------------------------------
// Implicit-GEMM 3x3 conv (pad=1) over NHWC f16 activations using WMMA.
//   out[p, co] = bias[co] + sum_k A[p, k] * Wpk[co, k]
//   k = (ky*3+kx)*Cin + ci   (Cin multiple of 32 => each 32-chunk is one tap)
// M = B*H*W pixels (W multiple of 16 => a 16-row tile = 16 pixels of one row).
// ---------------------------------------------------------------------------
__global__ void conv_gemm_wmma(const _Float16* __restrict__ in,   // (B,H,W,Cin)
                               const _Float16* __restrict__ Wpk,  // (Cout,9*Cin)
                               const float* __restrict__ bias,
                               float* __restrict__ out,           // (B,H,W,Cout)
                               int H, int Wd, int Cin, int Cout, int ntiles) {
  int wave = blockIdx.x * (blockDim.x >> 5) + (threadIdx.x >> 5);
  if (wave >= ntiles) return;
  int lane = threadIdx.x & 31;
  int ntN = Cout >> 4;
  int pix0 = (wave / ntN) << 4;
  int n0   = (wave % ntN) << 4;
  int r  = lane & 15;
  int hi = lane >> 4;
  int K = 9 * Cin;

  int x0   = pix0 % Wd;
  int rest = pix0 / Wd;
  int y = rest % H;
  int b = rest / H;
  int x = x0 + r;

  const _Float16* bp  = Wpk + (size_t)(n0 + r) * K + hi * 8;
  const _Float16* inb = in + (size_t)b * H * Wd * Cin;

  v8f  acc = {};
  v16h zf  = {};
  for (int k0 = 0; k0 < K; k0 += 32) {
    int f   = k0 / Cin;
    int ci0 = k0 % Cin;
    int ky = f / 3, kx = f % 3;
    int yy = y + ky - 1;
    int xx = x + kx - 1;
    union Frag { v16h v; h8v h[2]; };
    Frag fa, fb;
    if (yy >= 0 && yy < H && xx >= 0 && xx < Wd) {
      const _Float16* ap = inb + ((size_t)yy * Wd + xx) * Cin + ci0 + hi * 8;
      fa.h[0] = *(const h8v*)(ap);
      fa.h[1] = *(const h8v*)(ap + 16);
    } else {
      fa.v = zf;  // zero padding tap
    }
    fb.h[0] = *(const h8v*)(bp + k0);
    fb.h[1] = *(const h8v*)(bp + k0 + 16);
    acc = __builtin_amdgcn_wmma_f32_16x16x32_f16(
        false, fa.v, false, fb.v, (short)0, acc, false, false);
  }

  int co = n0 + r;
  float bv = bias[co];
#pragma unroll
  for (int j = 0; j < 8; ++j) {
    int p = pix0 + j + (hi << 3);
    out[(size_t)p * Cout + co] = acc[j] + bv;
  }
}

// Pack OIHW f32 conv weights -> (Cout, 9*Cin) f16, k = (ky*3+kx)*Cin + ci
__global__ void pack_conv_w_kernel(const float* __restrict__ w,
                                   _Float16* __restrict__ wp,
                                   int Cin, int Cout) {
  int K = 9 * Cin;
  int idx = blockIdx.x * blockDim.x + threadIdx.x;
  if (idx >= Cout * K) return;
  int co = idx / K;
  int k  = idx % K;
  int f  = k / Cin;
  int ci = k % Cin;
  int ky = f / 3, kx = f % 3;
  wp[idx] = (_Float16)w[((size_t)(co * Cin + ci) * 3 + ky) * 3 + kx];
}

// Direct conv0 (Cin=1), NCHW input -> NHWC f32 output (B,H,W,32)
__global__ void conv0_nhwc_kernel(const float* __restrict__ x,
                                  const float* __restrict__ w,   // (32,1,3,3)
                                  const float* __restrict__ bias,
                                  float* __restrict__ out,
                                  int H, int Wd) {
  size_t idx = (size_t)blockIdx.x * blockDim.x + threadIdx.x;
  size_t total = (size_t)BSZ * H * Wd * 32;
  if (idx >= total) return;
  int co = (int)(idx & 31);
  size_t p = idx >> 5;
  int xx0 = (int)(p % Wd); size_t t = p / Wd;
  int y = (int)(t % H);
  int b = (int)(t / H);
  const float* ip = x + (size_t)b * H * Wd;
  float acc = bias[co];
#pragma unroll
  for (int ky = 0; ky < 3; ++ky) {
    int yy = y + ky - 1;
    if (yy < 0 || yy >= H) continue;
#pragma unroll
    for (int kx = 0; kx < 3; ++kx) {
      int xx = xx0 + kx - 1;
      if (xx < 0 || xx >= Wd) continue;
      acc += w[co * 9 + ky * 3 + kx] * ip[(size_t)yy * Wd + xx];
    }
  }
  out[idx] = acc;
}

// Per-channel batch statistics over NHWC (training-mode BN), block per channel.
__global__ void bn_stats_nhwc_kernel(const float* __restrict__ x,
                                     float* __restrict__ mean,
                                     float* __restrict__ var,
                                     int C, size_t P) {
  int c = blockIdx.x;
  int tid = threadIdx.x;
  float s = 0.f, s2 = 0.f;
  for (size_t p = tid; p < P; p += blockDim.x) {
    float v = x[p * C + c];
    s += v; s2 += v * v;
  }
  __shared__ float rs[256], rs2[256];
  rs[tid] = s; rs2[tid] = s2; __syncthreads();
  for (int o = 128; o > 0; o >>= 1) {
    if (tid < o) { rs[tid] += rs[tid + o]; rs2[tid] += rs2[tid + o]; }
    __syncthreads();
  }
  if (tid == 0) {
    float fn = (float)P;
    float m = rs[0] / fn;
    mean[c] = m;
    var[c] = rs2[0] / fn - m * m;
  }
}

// BN affine + ReLU + 2x2 maxpool over NHWC; emits f32 and f16 copies.
__global__ void bn_relu_pool_nhwc_kernel(const float* __restrict__ xin,
                                         const float* __restrict__ mean,
                                         const float* __restrict__ var,
                                         const float* __restrict__ gamma,
                                         const float* __restrict__ beta,
                                         float* __restrict__ outf,
                                         _Float16* __restrict__ outh,
                                         int H, int Wd, int C) {
  int OH = H >> 1, OW = Wd >> 1;
  size_t idx = (size_t)blockIdx.x * blockDim.x + threadIdx.x;
  size_t total = (size_t)BSZ * OH * OW * C;
  if (idx >= total) return;
  int c = (int)(idx % C); size_t t = idx / C;
  int ox = (int)(t % OW); t /= OW;
  int oy = (int)(t % OH);
  int b  = (int)(t / OH);

  float m = mean[c];
  float inv = rsqrtf(var[c] + 1e-5f);
  float g = gamma[c], bt = beta[c];
  float best = 0.f;  // relu floor
#pragma unroll
  for (int dy = 0; dy < 2; ++dy)
#pragma unroll
    for (int dx = 0; dx < 2; ++dx) {
      float v = xin[(((size_t)b * H + (2 * oy + dy)) * Wd + (2 * ox + dx)) * C + c];
      v = (v - m) * inv * g + bt;
      best = fmaxf(best, v);
    }
  outf[idx] = best;
  outh[idx] = (_Float16)best;
}

// pooled conv3 NHWC (B,8,32,128) == (B,L,D); add 2-D PE -> feat f32 + f16
__global__ void feat_pe_kernel(const float* __restrict__ conv,
                               float* __restrict__ feat,
                               _Float16* __restrict__ feat16) {
  int idx = blockIdx.x * blockDim.x + threadIdx.x;  // B*L*D = 1048576
  if (idx >= BSZ * LLEN * DIM) return;
  int d = idx & (DIM - 1);
  int l = (idx >> 7) & (LLEN - 1);
  int hh = l >> 5, ww = l & 31;
  float v = conv[idx];
  int i = (d & 63) >> 1;
  float freq = __powf(10000.0f, (float)i / 32.0f);
  float arg = freq * (float)((d < 64) ? hh : ww);
  float pe = (d & 1) ? __cosf(arg) : __sinf(arg);
  v += pe;
  feat[idx] = v;
  feat16[idx] = (_Float16)v;
}

// Per-step: ey = emb[target[b, s+1]] (f32 + f16), h16 = f16(h)
__global__ void step_prepare_kernel(const int* __restrict__ target,
                                    const float* __restrict__ emb,
                                    int s,
                                    float* __restrict__ ey,
                                    _Float16* __restrict__ ey16,
                                    const float* __restrict__ h,
                                    _Float16* __restrict__ h16) {
  int idx = blockIdx.x * blockDim.x + threadIdx.x;  // B*E = 8192
  if (idx >= BSZ * EMB) return;
  int b = idx >> 8;
  int j = idx & 255;
  int y = target[b * MAXLEN + (s + 1)];
  float v = emb[(size_t)y * EMB + j];
  ey[idx] = v;
  ey16[idx] = (_Float16)v;
  h16[idx] = (_Float16)h[idx];
}

// Attention: e = tanh(ua + wa[:,None,:]) @ nu ; alpha = softmax; ct = alpha@feat
__global__ void attention_kernel(const float* __restrict__ ua,
                                 const float* __restrict__ wa,
                                 const float* __restrict__ nu,
                                 const float* __restrict__ feat,
                                 float* __restrict__ ct,
                                 _Float16* __restrict__ ct16) {
  int b = blockIdx.x;
  int tid = threadIdx.x;
  __shared__ float wa_s[LLEN], nu_s[LLEN], sm[LLEN], alpha[LLEN];
  wa_s[tid] = wa[b * LLEN + tid];
  nu_s[tid] = nu[tid];
  __syncthreads();

  const float* u = ua + ((size_t)b * LLEN + tid) * LLEN;
  float e = 0.f;
  for (int j = 0; j < LLEN; ++j) e += nu_s[j] * tanhf(u[j] + wa_s[j]);

  sm[tid] = e; __syncthreads();
  for (int o = 128; o > 0; o >>= 1) {
    if (tid < o) sm[tid] = fmaxf(sm[tid], sm[tid + o]);
    __syncthreads();
  }
  float mx = sm[0];
  __syncthreads();
  float ex = __expf(e - mx);
  sm[tid] = ex; __syncthreads();
  for (int o = 128; o > 0; o >>= 1) {
    if (tid < o) sm[tid] += sm[tid + o];
    __syncthreads();
  }
  float al = ex / sm[0];
  alpha[tid] = al;
  __syncthreads();

  if (tid < DIM) {
    float a = 0.f;
    for (int l = 0; l < LLEN; ++l)
      a += alpha[l] * feat[((size_t)b * LLEN + l) * DIM + tid];
    ct[b * DIM + tid] = a;
    ct16[b * DIM + tid] = (_Float16)a;
  }
}

// zr (B,512) pre-activations -> zt = sigmoid(z), rh16 = f16(sigmoid(r)*h)
__global__ void gates_kernel(const float* __restrict__ zr,
                             const float* __restrict__ h,
                             float* __restrict__ zt,
                             _Float16* __restrict__ rh16) {
  int idx = blockIdx.x * blockDim.x + threadIdx.x;  // B*HID
  if (idx >= BSZ * HID) return;
  int b = idx >> 8, j = idx & 255;
  float z = 1.f / (1.f + __expf(-zr[b * 2 * HID + j]));
  float r = 1.f / (1.f + __expf(-zr[b * 2 * HID + HID + j]));
  zt[idx] = z;
  rh16[idx] = (_Float16)(r * h[idx]);
}

// hn = (1-z)*h + z*tanh(hpre); h := hn; hn16 = f16(hn)
__global__ void hupdate_kernel(const float* __restrict__ hpre,
                               const float* __restrict__ zt,
                               float* __restrict__ h,
                               _Float16* __restrict__ hn16) {
  int idx = blockIdx.x * blockDim.x + threadIdx.x;
  if (idx >= BSZ * HID) return;
  float hti = tanhf(hpre[idx]);
  float z = zt[idx];
  float hn = (1.f - z) * h[idx] + z * hti;
  h[idx] = hn;
  hn16[idx] = (_Float16)hn;
}

// t1 += ey ; t116 = f16(t1)
__global__ void addey_kernel(float* __restrict__ t1,
                             const float* __restrict__ ey,
                             _Float16* __restrict__ t116) {
  int idx = blockIdx.x * blockDim.x + threadIdx.x;
  if (idx >= BSZ * EMB) return;
  float v = t1[idx] + ey[idx];
  t1[idx] = v;
  t116[idx] = (_Float16)v;
}

__global__ void f32_to_f16_kernel(const float* __restrict__ in,
                                  _Float16* __restrict__ out, int n) {
  int i = blockIdx.x * blockDim.x + threadIdx.x;
  if (i < n) out[i] = (_Float16)in[i];
}

__global__ void copy_f32_kernel(const float* __restrict__ in,
                                float* __restrict__ out, int n) {
  int i = blockIdx.x * blockDim.x + threadIdx.x;
  if (i < n) out[i] = in[i];
}

__global__ void fill_f32_kernel(float* __restrict__ p, float v, int n) {
  int i = blockIdx.x * blockDim.x + threadIdx.x;
  if (i < n) p[i] = v;
}

// out[b, 0, :] = 2.0
__global__ void init_out0_kernel(float* __restrict__ out) {
  int idx = blockIdx.x * blockDim.x + threadIdx.x;  // B*V
  if (idx >= BSZ * VOC) return;
  int b = idx >> 9, v = idx & 511;
  out[(size_t)b * MAXLEN * VOC + v] = 2.0f;
}

// ---------------------------------------------------------------------------
// Host side
// ---------------------------------------------------------------------------
static inline size_t align_up(size_t x, size_t a) { return (x + a - 1) & ~(a - 1); }

static inline void launch_gemm(hipStream_t st, const _Float16* A, const _Float16* Wt,
                               const float* bias, float* C,
                               int M, int N, int K, int ldc, int acc) {
  int tiles = (M / 16) * (N / 16);
  const int wpb = 4;  // 4 waves (128 threads) per block
  int blocks = CDIV(tiles, wpb);
  gemm_wmma_f16<<<dim3(blocks), dim3(32 * wpb), 0, st>>>(A, Wt, bias, C, M, N, K,
                                                         ldc, acc, tiles);
}

static inline void launch_conv_gemm(hipStream_t st, const _Float16* in,
                                    const _Float16* Wpk, const float* bias,
                                    float* out, int H, int Wd, int Cin, int Cout) {
  int tiles = ((BSZ * H * Wd) / 16) * (Cout / 16);
  const int wpb = 4;
  int blocks = CDIV(tiles, wpb);
  conv_gemm_wmma<<<dim3(blocks), dim3(32 * wpb), 0, st>>>(in, Wpk, bias, out,
                                                          H, Wd, Cin, Cout, tiles);
}

static inline void cvt16(hipStream_t st, const float* in, _Float16* out, int n) {
  f32_to_f16_kernel<<<CDIV(n, 256), 256, 0, st>>>(in, out, n);
}

extern "C" void kernel_launch(void* const* d_in, const int* in_sizes, int n_in,
                              void* d_out, int out_size, void* d_ws, size_t ws_size,
                              hipStream_t stream) {
  (void)in_sizes; (void)n_in; (void)out_size; (void)ws_size;

  const float* x      = (const float*)d_in[IN_X];
  const int*   target = (const int*)d_in[IN_TARGET];
  const float* emb    = (const float*)d_in[IN_emb];
  const float* nu     = (const float*)d_in[IN_nu_w];
  float* out = (float*)d_out;

  // ---- workspace carve-up --------------------------------------------------
  char* base = (char*)d_ws;
  size_t off = 0;
  auto carve = [&](size_t bytes) {
    char* p = base + off;
    off = align_up(off + bytes, 256);
    return p;
  };
  float* raw  = (float*)carve((size_t)32 * 128 * 512 * 32 * 4);   // 268 MB (max raw NHWC)
  float* p0f  = (float*)carve((size_t)32 * 64 * 256 * 32 * 4);    // 67 MB
  _Float16* p0h = (_Float16*)carve((size_t)32 * 64 * 256 * 32 * 2);
  float* p1f  = (float*)carve((size_t)32 * 32 * 128 * 64 * 4);    // 34 MB
  _Float16* p1h = (_Float16*)carve((size_t)32 * 32 * 128 * 64 * 2);
  float* meanb  = (float*)carve(512);
  float* varb   = (float*)carve(512);
  float* feat   = (float*)carve((size_t)BSZ * LLEN * DIM * 4);
  _Float16* feat16 = (_Float16*)carve((size_t)BSZ * LLEN * DIM * 2);
  float* ua     = (float*)carve((size_t)BSZ * LLEN * LLEN * 4);   // 8 MB

  _Float16* wpk1 = (_Float16*)carve((size_t)64 * 9 * 32 * 2);
  _Float16* wpk2 = (_Float16*)carve((size_t)64 * 9 * 64 * 2);
  _Float16* wpk3 = (_Float16*)carve((size_t)128 * 9 * 64 * 2);

  _Float16* Ua16   = (_Float16*)carve(LLEN * DIM * 2);
  _Float16* Wa16   = (_Float16*)carve(LLEN * HID * 2);
  _Float16* Wzry16 = (_Float16*)carve(2 * HID * EMB * 2);
  _Float16* Uzr16  = (_Float16*)carve(2 * HID * HID * 2);
  _Float16* Czr16  = (_Float16*)carve(2 * HID * DIM * 2);
  _Float16* Wyh16  = (_Float16*)carve(HID * EMB * 2);
  _Float16* Urh16  = (_Float16*)carve(HID * HID * 2);
  _Float16* Wc16   = (_Float16*)carve(EMB * DIM * 2);
  _Float16* Wh16   = (_Float16*)carve(EMB * HID * 2);
  _Float16* Wo16   = (_Float16*)carve(VOC * EMB * 2);
  float* bzr_y = (float*)carve(2 * HID * 4);
  float* bzr_u = (float*)carve(2 * HID * 4);
  float* bzr_c = (float*)carve(2 * HID * 4);

  float* h   = (float*)carve(BSZ * HID * 4);
  _Float16* h16 = (_Float16*)carve(BSZ * HID * 2);
  float* ey  = (float*)carve(BSZ * EMB * 4);
  _Float16* ey16 = (_Float16*)carve(BSZ * EMB * 2);
  float* wa  = (float*)carve(BSZ * LLEN * 4);
  float* ct  = (float*)carve(BSZ * DIM * 4);
  _Float16* ct16 = (_Float16*)carve(BSZ * DIM * 2);
  float* zr  = (float*)carve(BSZ * 2 * HID * 4);
  float* zt  = (float*)carve(BSZ * HID * 4);
  _Float16* rh16 = (_Float16*)carve(BSZ * HID * 2);
  float* hpre = (float*)carve(BSZ * HID * 4);
  _Float16* hn16 = (_Float16*)carve(BSZ * HID * 2);
  float* t1  = (float*)carve(BSZ * EMB * 4);
  _Float16* t116 = (_Float16*)carve(BSZ * EMB * 2);

  // ---- one-time weight conversion / packing --------------------------------
  pack_conv_w_kernel<<<CDIV(64 * 9 * 32, 256), 256, 0, stream>>>(
      (const float*)d_in[IN_c1_w], wpk1, 32, 64);
  pack_conv_w_kernel<<<CDIV(64 * 9 * 64, 256), 256, 0, stream>>>(
      (const float*)d_in[IN_c2_w], wpk2, 64, 64);
  pack_conv_w_kernel<<<CDIV(128 * 9 * 64, 256), 256, 0, stream>>>(
      (const float*)d_in[IN_c3_w], wpk3, 64, 128);

  cvt16(stream, (const float*)d_in[IN_Ua_w], Ua16, LLEN * DIM);
  cvt16(stream, (const float*)d_in[IN_Wa_w], Wa16, LLEN * HID);
  cvt16(stream, (const float*)d_in[IN_Wyz_w], Wzry16, HID * EMB);
  cvt16(stream, (const float*)d_in[IN_Wyr_w], Wzry16 + HID * EMB, HID * EMB);
  cvt16(stream, (const float*)d_in[IN_Uhz_w], Uzr16, HID * HID);
  cvt16(stream, (const float*)d_in[IN_Uhr_w], Uzr16 + HID * HID, HID * HID);
  cvt16(stream, (const float*)d_in[IN_Ccz_w], Czr16, HID * DIM);
  cvt16(stream, (const float*)d_in[IN_Ccr_w], Czr16 + HID * DIM, HID * DIM);
  cvt16(stream, (const float*)d_in[IN_Wyh_w], Wyh16, HID * EMB);
  cvt16(stream, (const float*)d_in[IN_Urh_w], Urh16, HID * HID);
  cvt16(stream, (const float*)d_in[IN_Wc_w], Wc16, EMB * DIM);
  cvt16(stream, (const float*)d_in[IN_Wh_w], Wh16, EMB * HID);
  cvt16(stream, (const float*)d_in[IN_Wo_w], Wo16, VOC * EMB);
  copy_f32_kernel<<<1, 256, 0, stream>>>((const float*)d_in[IN_Wyz_b], bzr_y, HID);
  copy_f32_kernel<<<1, 256, 0, stream>>>((const float*)d_in[IN_Wyr_b], bzr_y + HID, HID);
  copy_f32_kernel<<<1, 256, 0, stream>>>((const float*)d_in[IN_Uhz_b], bzr_u, HID);
  copy_f32_kernel<<<1, 256, 0, stream>>>((const float*)d_in[IN_Uhr_b], bzr_u + HID, HID);
  copy_f32_kernel<<<1, 256, 0, stream>>>((const float*)d_in[IN_Ccz_b], bzr_c, HID);
  copy_f32_kernel<<<1, 256, 0, stream>>>((const float*)d_in[IN_Ccr_b], bzr_c + HID, HID);

  // ---- watcher (NHWC): conv -> BN(batch stats) -> relu -> 2x2 maxpool -----
  // L0: direct conv (Cin=1) -> raw (B,128,512,32)
  {
    size_t total = (size_t)BSZ * 128 * 512 * 32;
    conv0_nhwc_kernel<<<(unsigned)CDIV(total, 256), 256, 0, stream>>>(
        x, (const float*)d_in[IN_c0_w], (const float*)d_in[IN_c0_b], raw, 128, 512);
    bn_stats_nhwc_kernel<<<32, 256, 0, stream>>>(raw, meanb, varb, 32,
                                                 (size_t)BSZ * 128 * 512);
    bn_relu_pool_nhwc_kernel<<<(unsigned)CDIV(total / 4, 256), 256, 0, stream>>>(
        raw, meanb, varb, (const float*)d_in[IN_c0_gamma],
        (const float*)d_in[IN_c0_beta], p0f, p0h, 128, 512, 32);
  }
  // L1: implicit-GEMM WMMA conv (B,64,256,32) -> raw (B,64,256,64)
  {
    launch_conv_gemm(stream, p0h, wpk1, (const float*)d_in[IN_c1_b], raw,
                     64, 256, 32, 64);
    bn_stats_nhwc_kernel<<<64, 256, 0, stream>>>(raw, meanb, varb, 64,
                                                 (size_t)BSZ * 64 * 256);
    size_t ptotal = (size_t)BSZ * 32 * 128 * 64;
    bn_relu_pool_nhwc_kernel<<<(unsigned)CDIV(ptotal, 256), 256, 0, stream>>>(
        raw, meanb, varb, (const float*)d_in[IN_c1_gamma],
        (const float*)d_in[IN_c1_beta], p1f, p1h, 64, 256, 64);
  }
  // L2: (B,32,128,64) -> raw (B,32,128,64)
  {
    launch_conv_gemm(stream, p1h, wpk2, (const float*)d_in[IN_c2_b], raw,
                     32, 128, 64, 64);
    bn_stats_nhwc_kernel<<<64, 256, 0, stream>>>(raw, meanb, varb, 64,
                                                 (size_t)BSZ * 32 * 128);
    size_t ptotal = (size_t)BSZ * 16 * 64 * 64;
    bn_relu_pool_nhwc_kernel<<<(unsigned)CDIV(ptotal, 256), 256, 0, stream>>>(
        raw, meanb, varb, (const float*)d_in[IN_c2_gamma],
        (const float*)d_in[IN_c2_beta], p0f, p0h, 32, 128, 64);
  }
  // L3: (B,16,64,64) -> raw (B,16,64,128)
  {
    launch_conv_gemm(stream, p0h, wpk3, (const float*)d_in[IN_c3_b], raw,
                     16, 64, 64, 128);
    bn_stats_nhwc_kernel<<<128, 256, 0, stream>>>(raw, meanb, varb, 128,
                                                  (size_t)BSZ * 16 * 64);
    size_t ptotal = (size_t)BSZ * 8 * 32 * 128;
    bn_relu_pool_nhwc_kernel<<<(unsigned)CDIV(ptotal, 256), 256, 0, stream>>>(
        raw, meanb, varb, (const float*)d_in[IN_c3_gamma],
        (const float*)d_in[IN_c3_beta], p1f, p1h, 16, 64, 128);
  }

  // ---- feat = conv_out + PE ; ua = feat @ U_a^T + b -----------------------
  feat_pe_kernel<<<CDIV(BSZ * LLEN * DIM, 256), 256, 0, stream>>>(p1f, feat, feat16);
  launch_gemm(stream, feat16, Ua16, (const float*)d_in[IN_Ua_b], ua,
              BSZ * LLEN, LLEN, DIM, LLEN, 0);

  // ---- decoder init -------------------------------------------------------
  fill_f32_kernel<<<CDIV(BSZ * HID, 256), 256, 0, stream>>>(h, 0.0f, BSZ * HID);
  init_out0_kernel<<<CDIV(BSZ * VOC, 256), 256, 0, stream>>>(out);

  // ---- 127 sequential decode steps (teacher forcing) ----------------------
  for (int s = 0; s < MAXLEN - 1; ++s) {
    step_prepare_kernel<<<CDIV(BSZ * EMB, 256), 256, 0, stream>>>(
        target, emb, s, ey, ey16, h, h16);

    // wa = h @ W_a^T + b
    launch_gemm(stream, h16, Wa16, (const float*)d_in[IN_Wa_b], wa,
                BSZ, LLEN, HID, LLEN, 0);

    attention_kernel<<<BSZ, LLEN, 0, stream>>>(ua, wa, nu, feat, ct, ct16);

    // zr = ey@[Wyz;Wyr]^T + h@[Uhz;Uhr]^T + ct@[Ccz;Ccr]^T (+ all biases)
    launch_gemm(stream, ey16, Wzry16, bzr_y, zr, BSZ, 2 * HID, EMB, 2 * HID, 0);
    launch_gemm(stream, h16,  Uzr16,  bzr_u, zr, BSZ, 2 * HID, HID, 2 * HID, 1);
    launch_gemm(stream, ct16, Czr16,  bzr_c, zr, BSZ, 2 * HID, DIM, 2 * HID, 1);
    gates_kernel<<<CDIV(BSZ * HID, 256), 256, 0, stream>>>(zr, h, zt, rh16);

    // hpre = ey@Wyh^T + (rt*h)@Urh^T + ct@Ccz^T  (C_cz reuse is faithful)
    launch_gemm(stream, ey16, Wyh16, (const float*)d_in[IN_Wyh_b], hpre,
                BSZ, HID, EMB, HID, 0);
    launch_gemm(stream, rh16, Urh16, (const float*)d_in[IN_Urh_b], hpre,
                BSZ, HID, HID, HID, 1);
    launch_gemm(stream, ct16, Czr16 /* rows 0..255 == C_cz */,
                (const float*)d_in[IN_Ccz_b], hpre, BSZ, HID, DIM, HID, 1);
    hupdate_kernel<<<CDIV(BSZ * HID, 256), 256, 0, stream>>>(hpre, zt, h, hn16);

    // t1 = ct@Wc^T + hn@Wh^T + ey ; o = t1@Wo^T + b -> out[:, s+1, :]
    launch_gemm(stream, ct16, Wc16, (const float*)d_in[IN_Wc_b], t1,
                BSZ, EMB, DIM, EMB, 0);
    launch_gemm(stream, hn16, Wh16, (const float*)d_in[IN_Wh_b], t1,
                BSZ, EMB, HID, EMB, 1);
    addey_kernel<<<CDIV(BSZ * EMB, 256), 256, 0, stream>>>(t1, ey, t116);
    launch_gemm(stream, t116, Wo16, (const float*)d_in[IN_Wo_b],
                out + (size_t)(s + 1) * VOC, BSZ, VOC, EMB, MAXLEN * VOC, 0);
  }
}
